// TypeInferModel_36610301231302
// MI455X (gfx1250) — matrix operand order repaired and checked by reading, not compiled, not run on hardware
//
#include <hip/hip_runtime.h>
#include <hip/hip_bf16.h>

typedef __bf16 bf16;
typedef __attribute__((ext_vector_type(16))) __bf16 v16bf;
typedef __attribute__((ext_vector_type(8)))  float  v8f;
typedef int v4i __attribute__((__vector_size__(16)));

#define MBLK 64
#define NBLK 256
#define KS   32
#define LDA  40   // halves; byte stride 80 keeps 16B alignment for 8-half chunks
#define LDB  40

#define AS1 __attribute__((address_space(1)))
#define AS3 __attribute__((address_space(3)))

#if defined(__HIP_DEVICE_COMPILE__) && __has_builtin(__builtin_amdgcn_global_load_async_to_lds_b128)
#define HAVE_ASYNC_LDS 1
#else
#define HAVE_ASYNC_LDS 0
#endif

__device__ __forceinline__ void wait_async() {
#if HAVE_ASYNC_LDS
#if __has_builtin(__builtin_amdgcn_s_wait_asynccnt)
  __builtin_amdgcn_s_wait_asynccnt(0);
#else
  asm volatile("s_wait_asynccnt 0x0" ::: "memory");
#endif
#endif
}

__device__ __forceinline__ float lrelu(float x) { return x > 0.0f ? x : 0.01f * x; }

// ---------------- small utility kernels ----------------

__global__ void k_fill(float* __restrict__ p, float v, int n) {
  int i = blockIdx.x * blockDim.x + threadIdx.x;
  if (i < n) p[i] = v;
}

__global__ void k_deg(const int* __restrict__ dst, float* __restrict__ deg, int E) {
  int i = blockIdx.x * blockDim.x + threadIdx.x;
  if (i < E) unsafeAtomicAdd(&deg[dst[i]], 1.0f);
}

__global__ void k_rsqrt_ip(float* __restrict__ p, int n) {
  int i = blockIdx.x * blockDim.x + threadIdx.x;
  if (i < n) p[i] = rsqrtf(p[i]);
}

__global__ void k_cast_bf16(const float* __restrict__ a, bf16* __restrict__ b, int n) {
  int i = blockIdx.x * blockDim.x + threadIdx.x;
  if (i < n) b[i] = (bf16)a[i];
}

// h[node, f] = (bf16) emb[x[node], f], 128 features
__global__ void k_embed(const int* __restrict__ x, const float* __restrict__ emb,
                        bf16* __restrict__ h, int n /* = N*128 */) {
  int i = blockIdx.x * blockDim.x + threadIdx.x;
  if (i < n) {
    int node = i >> 7;
    int f = i & 127;
    h[i] = (bf16)emb[(size_t)x[node] * 128 + f];
  }
}

// out[i,f] = t[i,f]*dinv[i]^2 + bias[f]   (self-loop term + conv bias)
__global__ void k_agg_init(const float* __restrict__ t, const float* __restrict__ dinv,
                           const float* __restrict__ bias, float* __restrict__ out,
                           long nf, int lgF, int fmask) {
  long i = (long)blockIdx.x * blockDim.x + threadIdx.x;
  if (i < nf) {
    int node = (int)(i >> lgF);
    int f = (int)(i & fmask);
    float di = dinv[node];
    out[i] = t[i] * di * di + bias[f];
  }
}

// out[dst, f..f+3] += t[src, f..f+3] * dinv[src]*dinv[dst]
__global__ void k_agg_edges(const float* __restrict__ t, const float* __restrict__ dinv,
                            const int* __restrict__ src, const int* __restrict__ dst,
                            float* __restrict__ out, long nwork, int lgVPE, int F) {
  long i = (long)blockIdx.x * blockDim.x + threadIdx.x;
  if (i < nwork) {
    int e = (int)(i >> lgVPE);
    int f4 = (int)((i & ((1 << lgVPE) - 1)) << 2);
    int s = src[e], d = dst[e];
    float w = dinv[s] * dinv[d];
    const float4 v = *(const float4*)(t + (size_t)s * F + f4);
    float* o = out + (size_t)d * F + f4;
    unsafeAtomicAdd(o + 0, v.x * w);
    unsafeAtomicAdd(o + 1, v.y * w);
    unsafeAtomicAdd(o + 2, v.z * w);
    unsafeAtomicAdd(o + 3, v.w * w);
  }
}

// column sums / sum-of-squares with LDS float atomics, then global atomics
__global__ __launch_bounds__(256)
void k_bn_stats(const float* __restrict__ y, float* __restrict__ gsum,
                float* __restrict__ gsq, long nf, int F, int fmask) {
  __shared__ float ls[512];
  __shared__ float lq[512];
  for (int c = threadIdx.x; c < F; c += 256) { ls[c] = 0.f; lq[c] = 0.f; }
  __syncthreads();
  for (long i = (long)blockIdx.x * 256 + threadIdx.x; i < nf; i += (long)gridDim.x * 256) {
    float v = y[i];
    int c = (int)(i & fmask);
    atomicAdd(&ls[c], v);       // ds_add_f32
    atomicAdd(&lq[c], v * v);
  }
  __syncthreads();
  for (int c = threadIdx.x; c < F; c += 256) {
    unsafeAtomicAdd(&gsum[c], ls[c]);
    unsafeAtomicAdd(&gsq[c], lq[c]);
  }
}

__global__ void k_bn_coeff(const float* __restrict__ gsum, const float* __restrict__ gsq,
                           const float* __restrict__ gamma, const float* __restrict__ beta,
                           float* __restrict__ scale, float* __restrict__ shift,
                           float invN, int F) {
  int c = blockIdx.x * blockDim.x + threadIdx.x;
  if (c < F) {
    float mean = gsum[c] * invN;
    float var = gsq[c] * invN - mean * mean;
    float s = gamma[c] * rsqrtf(var + 1e-5f);
    scale[c] = s;
    shift[c] = beta[c] - mean * s;
  }
}

// ybf = bf16( lrelu( y*scale + shift ) )
__global__ void k_bn_apply(const float* __restrict__ y, const float* __restrict__ scale,
                           const float* __restrict__ shift, bf16* __restrict__ o,
                           long nf, int fmask) {
  long i = (long)blockIdx.x * blockDim.x + threadIdx.x;
  if (i < nf) {
    int c = (int)(i & fmask);
    o[i] = (bf16)lrelu(y[i] * scale[c] + shift[c]);
  }
}

__global__ void k_out_init(float* __restrict__ out, const float* __restrict__ b4, int n) {
  int i = blockIdx.x * blockDim.x + threadIdx.x;
  if (i < n) out[i] = b4[i % 6];
}

// ---------------- shared WMMA mainloop ----------------
// block = 256 threads = 8 waves (2 M x 4 N), wave tile 32x64, block tile 64x256.
// A staged row-major (async DMA to LDS when available), B staged transposed.

__device__ __forceinline__ void wmma_mainloop(const bf16* __restrict__ A,
                                              const bf16* __restrict__ B,
                                              bf16* sA, bf16* sB, v8f (&acc)[2][4],
                                              int M, int N, int K, int m0, int n0) {
  const int tid = threadIdx.x;
  const int lane = tid & 31, wave = tid >> 5;
  const int wm = wave >> 2, wn = wave & 3;
  const int lm = lane & 15, lh = lane >> 4;

  for (int k0 = 0; k0 < K; k0 += KS) {
    // A tile 64x32: exactly one 16B chunk per thread, row-guarded.
    // (rows >= M hold stale LDS data; their C rows are never stored.)
    {
      int m = tid >> 2, k = (tid & 3) << 3;
      int gm = m0 + m;
      if (gm < M) {
        const bf16* gsrc = A + (size_t)gm * K + k0 + k;
        bf16* ldst = sA + m * LDA + k;
#if HAVE_ASYNC_LDS
        __builtin_amdgcn_global_load_async_to_lds_b128((AS1 v4i*)gsrc,
                                                       (AS3 v4i*)ldst, 0, 0);
#else
        *(uint4*)ldst = *(const uint4*)gsrc;
#endif
      }
    }
    // B tile 32x256 -> transposed LDS [n][k]; coalesced b128 global reads.
    for (int c = tid; c < (KS * NBLK / 8); c += 256) {
      int k = c >> 5;
      int nc = (c & 31) << 3;
      union { uint4 v; bf16 h[8]; } u;
      u.v = *(const uint4*)(B + (size_t)(k0 + k) * N + n0 + nc);
#pragma unroll
      for (int j = 0; j < 8; j++) sB[(nc + j) * LDB + k] = u.h[j];
    }
    // prefetch next K tile toward L2 (global_prefetch_b8)
    if (k0 + KS < K) {
      int pk = tid >> 5;
      int pn = (tid & 31) << 3;
      __builtin_prefetch(B + (size_t)(k0 + KS + pk) * N + n0 + pn, 0, 1);
      if (tid < 64) {
        int gm = m0 + tid; if (gm >= M) gm = M - 1;
        __builtin_prefetch(A + (size_t)gm * K + k0 + KS, 0, 1);
      }
    }
    wait_async();
    __syncthreads();

    v16bf af[2], bfr[4];
#pragma unroll
    for (int fm = 0; fm < 2; fm++) {
      int m = wm * 32 + fm * 16 + lm;
      int ks = lh * 8;
#pragma unroll
      for (int i = 0; i < 8; i++) af[fm][i] = sA[m * LDA + ks + i];
#pragma unroll
      for (int i = 0; i < 8; i++) af[fm][8 + i] = sA[m * LDA + 16 + ks + i];
    }
#pragma unroll
    for (int fn = 0; fn < 4; fn++) {
      int n = wn * 64 + fn * 16 + lm;
      int kb = lh * 8;
#pragma unroll
      for (int i = 0; i < 8; i++) bfr[fn][i] = sB[n * LDB + kb + i];
#pragma unroll
      for (int i = 0; i < 8; i++) bfr[fn][8 + i] = sB[n * LDB + 16 + kb + i];
    }
#pragma unroll
    for (int fm = 0; fm < 2; fm++)
#pragma unroll
      for (int fn = 0; fn < 4; fn++)
        acc[fm][fn] = __builtin_amdgcn_wmma_f32_16x16x32_bf16(
            false, af[fm], false, bfr[fn], (short)0, acc[fm][fn], false, false);
    __syncthreads();
  }
}

// ---------------- GEMM: C[M,N] = A[M,K] @ B[K,N] (f32 out) ----------------

__global__ __launch_bounds__(256)
void k_gemm(const bf16* __restrict__ A, const bf16* __restrict__ B,
            float* __restrict__ C, int M, int N, int K) {
  __shared__ bf16 sA[MBLK * LDA];
  __shared__ bf16 sB[NBLK * LDB];
  const int tid = threadIdx.x;
  const int lane = tid & 31, wave = tid >> 5;
  const int wm = wave >> 2, wn = wave & 3;
  const int m0 = blockIdx.x * MBLK;
  const int n0 = blockIdx.y * NBLK;
  const int lm = lane & 15, lh = lane >> 4;

  v8f acc[2][4] = {};
  wmma_mainloop(A, B, sA, sB, acc, M, N, K, m0, n0);

#pragma unroll
  for (int fm = 0; fm < 2; fm++)
#pragma unroll
    for (int fn = 0; fn < 4; fn++) {
      int n = n0 + wn * 64 + fn * 16 + lm;
#pragma unroll
      for (int v = 0; v < 8; v++) {
        int m = m0 + wm * 32 + fm * 16 + lh * 8 + v;
        if (m < M) C[(size_t)m * N + n] = acc[fm][fn][v];
      }
    }
}

// ------- fused: Y = lrelu(A@W3 + b3); out += Y_tile @ W4_slice -------
// Avoids materializing the [N,4096] activation (saves ~820MB of HBM traffic).

__global__ __launch_bounds__(256)
void k_gemm3_out(const bf16* __restrict__ A, const bf16* __restrict__ B,
                 const float* __restrict__ bias, const float* __restrict__ W4,
                 float* __restrict__ out, int M, int N, int K) {
  __shared__ bf16 sA[MBLK * LDA];
  __shared__ bf16 sB[NBLK * LDB];
  __shared__ bf16 sY[MBLK * NBLK];     // 32 KB
  __shared__ float sW4[NBLK * 6];      // 6 KB
  const int tid = threadIdx.x;
  const int lane = tid & 31, wave = tid >> 5;
  const int wm = wave >> 2, wn = wave & 3;
  const int m0 = blockIdx.x * MBLK;
  const int n0 = blockIdx.y * NBLK;
  const int lm = lane & 15, lh = lane >> 4;

  v8f acc[2][4] = {};
  wmma_mainloop(A, B, sA, sB, acc, M, N, K, m0, n0);

  // bias + leaky-relu into LDS tile
#pragma unroll
  for (int fm = 0; fm < 2; fm++)
#pragma unroll
    for (int fn = 0; fn < 4; fn++) {
      int nl = wn * 64 + fn * 16 + lm;
      float bv = bias[n0 + nl];
#pragma unroll
      for (int v = 0; v < 8; v++) {
        int ml = wm * 32 + fm * 16 + lh * 8 + v;
        sY[ml * NBLK + nl] = (bf16)lrelu(acc[fm][fn][v] + bv);
      }
    }
  for (int i = tid; i < NBLK * 6; i += 256) sW4[i] = W4[(size_t)n0 * 6 + i];
  __syncthreads();

  // out[m0+r, c] += sum_k sY[r,k] * sW4[k,c]
  for (int idx = tid; idx < MBLK * 6; idx += 256) {
    int r = idx / 6, c = idx - r * 6;
    int gm = m0 + r;
    if (gm < M) {
      float s = 0.f;
#pragma unroll 8
      for (int kk = 0; kk < NBLK; kk++)
        s += (float)sY[r * NBLK + kk] * sW4[kk * 6 + c];
      unsafeAtomicAdd(&out[(size_t)gm * 6 + c], s);
    }
  }
}

// ---------------- launcher ----------------

extern "C" void kernel_launch(void* const* d_in, const int* in_sizes, int n_in,
                              void* d_out, int out_size, void* d_ws, size_t ws_size,
                              hipStream_t stream) {
  const int*   x    = (const int*)d_in[0];
  const int*   a    = (const int*)d_in[1];
  const float* emb  = (const float*)d_in[2];
  const float* W1   = (const float*)d_in[3];
  const float* b1   = (const float*)d_in[4];
  const float* g1   = (const float*)d_in[5];
  const float* be1  = (const float*)d_in[6];
  const float* W2   = (const float*)d_in[7];
  const float* b2   = (const float*)d_in[8];
  const float* g2   = (const float*)d_in[9];
  const float* be2  = (const float*)d_in[10];
  const float* W3   = (const float*)d_in[11];
  const float* b3   = (const float*)d_in[12];
  const float* W4   = (const float*)d_in[13];
  const float* b4   = (const float*)d_in[14];
  float* out = (float*)d_out;

  const int N = in_sizes[0];            // 50000 nodes
  const int E = in_sizes[1] / 2;        // 1.6M edges
  const int* srcE = a;
  const int* dstE = a + E;

  // workspace carve-out
  char* w = (char*)d_ws;
  size_t off = 0;
  auto carve = [&](size_t bytes) -> void* {
    void* p = w + off;
    off += (bytes + 255) & ~(size_t)255;
    return p;
  };
  float* dinv  = (float*)carve((size_t)N * 4);
  float* gsum  = (float*)carve(512 * 4);
  float* gsq   = (float*)carve(512 * 4);
  float* scale = (float*)carve(512 * 4);
  float* shift = (float*)carve(512 * 4);
  bf16* w1b = (bf16*)carve((size_t)128 * 256 * 2);
  bf16* w2b = (bf16*)carve((size_t)256 * 512 * 2);
  bf16* w3b = (bf16*)carve((size_t)512 * 4096 * 2);
  float* tbuf = (float*)carve((size_t)N * 512 * 4);   // pre-aggregation h@W
  float* agg  = (float*)carve((size_t)N * 512 * 4);   // post-aggregation
  bf16* ybf   = (bf16*)carve((size_t)N * 512 * 2);    // bf16 activations (also holds h)

  auto cdiv = [](long n, long d) -> unsigned { return (unsigned)((n + d - 1) / d); };
  const unsigned mtiles = cdiv(N, MBLK);
  const float invN = 1.0f / (float)N;

  // degree / normalization
  k_fill<<<cdiv(N, 256), 256, 0, stream>>>(dinv, 1.0f, N);
  k_deg<<<cdiv(E, 256), 256, 0, stream>>>(dstE, dinv, E);
  k_rsqrt_ip<<<cdiv(N, 256), 256, 0, stream>>>(dinv, N);

  // weight down-converts
  k_cast_bf16<<<cdiv(128 * 256, 256), 256, 0, stream>>>(W1, w1b, 128 * 256);
  k_cast_bf16<<<cdiv(256 * 512, 256), 256, 0, stream>>>(W2, w2b, 256 * 512);
  k_cast_bf16<<<cdiv(512 * 4096, 256), 256, 0, stream>>>(W3, w3b, 512 * 4096);

  // embedding gather -> bf16 [N,128]
  k_embed<<<cdiv((long)N * 128, 256), 256, 0, stream>>>(x, emb, ybf, N * 128);

  // ---- layer 1: t = h @ W1 ; aggregate ; BN + lrelu -> ybf [N,256] ----
  k_gemm<<<dim3(mtiles, 1), 256, 0, stream>>>(ybf, w1b, tbuf, N, 256, 128);
  k_agg_init<<<cdiv((long)N * 256, 256), 256, 0, stream>>>(tbuf, dinv, b1, agg, (long)N * 256, 8, 255);
  k_agg_edges<<<cdiv((long)E * 64, 256), 256, 0, stream>>>(tbuf, dinv, srcE, dstE, agg, (long)E * 64, 6, 256);
  k_fill<<<2, 256, 0, stream>>>(gsum, 0.f, 512);
  k_fill<<<2, 256, 0, stream>>>(gsq, 0.f, 512);
  k_bn_stats<<<1024, 256, 0, stream>>>(agg, gsum, gsq, (long)N * 256, 256, 255);
  k_bn_coeff<<<2, 256, 0, stream>>>(gsum, gsq, g1, be1, scale, shift, invN, 256);
  k_bn_apply<<<cdiv((long)N * 256, 256), 256, 0, stream>>>(agg, scale, shift, ybf, (long)N * 256, 255);

  // ---- layer 2: t = y1 @ W2 ; aggregate ; BN + lrelu -> ybf [N,512] ----
  k_gemm<<<dim3(mtiles, 2), 256, 0, stream>>>(ybf, w2b, tbuf, N, 512, 256);
  k_agg_init<<<cdiv((long)N * 512, 256), 256, 0, stream>>>(tbuf, dinv, b2, agg, (long)N * 512, 9, 511);
  k_agg_edges<<<cdiv((long)E * 128, 256), 256, 0, stream>>>(tbuf, dinv, srcE, dstE, agg, (long)E * 128, 7, 512);
  k_fill<<<2, 256, 0, stream>>>(gsum, 0.f, 512);
  k_fill<<<2, 256, 0, stream>>>(gsq, 0.f, 512);
  k_bn_stats<<<1024, 256, 0, stream>>>(agg, gsum, gsq, (long)N * 512, 512, 511);
  k_bn_coeff<<<2, 256, 0, stream>>>(gsum, gsq, g2, be2, scale, shift, invN, 512);
  k_bn_apply<<<cdiv((long)N * 512, 256), 256, 0, stream>>>(agg, scale, shift, ybf, (long)N * 512, 511);

  // ---- layers 3+4 fused: out = lrelu(y2 @ W3 + b3) @ W4 + b4 ----
  k_out_init<<<cdiv((long)N * 6, 256), 256, 0, stream>>>(out, b4, N * 6);
  k_gemm3_out<<<dim3(mtiles, 16), 256, 0, stream>>>(ybf, w3b, b3, W4, out, N, 4096, 512);
}
